// W8A8B8O32Conv1D_32968168964175
// MI455X (gfx1250) — 8-bit matrix operand order repaired and checked by reading, compiled, not run on hardware
//
#include <hip/hip_runtime.h>

typedef __attribute__((ext_vector_type(8))) int v8i;
typedef __attribute__((ext_vector_type(4))) int v4i;

#define N_ROWS   32768
#define K_DIM    768
#define O_DIM    768
#define BM       128
#define BN       128
#define BK       64
#define A_STRIDE 72
#define B_STRIDE 80

static __device__ __forceinline__ int pack4(int x0, int x1, int x2, int x3) {
    unsigned t0 = __builtin_amdgcn_perm((unsigned)x1, (unsigned)x0, 0x00000400u);
    unsigned t1 = __builtin_amdgcn_perm((unsigned)x3, (unsigned)x2, 0x00000400u);
    return (int)__builtin_amdgcn_perm(t1, t0, 0x05040100u);
}

__global__ __launch_bounds__(256)
void w8a8_gemm_wmma(const int* __restrict__ x,
                    const int* __restrict__ w,
                    const _Float16* __restrict__ bias,
                    const float* __restrict__ alpha_p,
                    float* __restrict__ out) {
    __shared__ __align__(16) char ALds[2][BM * A_STRIDE];
    __shared__ __align__(16) char BLds[2][BN * B_STRIDE];

    const int tid  = threadIdx.x;
    const int wave = tid >> 5;
    const int lane = tid & 31;
    const int half = lane >> 4;
    const int l16  = lane & 15;

    const int mblk = blockIdx.x * BM;
    const int nblk = blockIdx.y * BN;

    const int mwave = wave >> 1;
    const int nwave = wave & 1;

    const int prow = tid >> 1;
    const int pkh  = (tid & 1) * 32;

    const float alpha = alpha_p[0];

    v8i acc[2][4];
#pragma unroll
    for (int i = 0; i < 2; ++i)
#pragma unroll
        for (int j = 0; j < 4; ++j) acc[i][j] = (v8i)0;

    v4i aReg[8];
    int bReg[32];

    auto load_global = [&](int k0) {
        const v4i* xg = (const v4i*)(x + (mblk + prow) * K_DIM + k0 + pkh);
#pragma unroll
        for (int j = 0; j < 8; ++j)
            aReg[j] = __builtin_nontemporal_load(xg + j);
        const int* wg = w + (k0 + pkh) * O_DIM + nblk + prow;
#pragma unroll
        for (int i = 0; i < 32; ++i) bReg[i] = wg[i * O_DIM];
    };

    auto pack_to_lds = [&](int buf) {
        char* arow = &ALds[buf][prow * A_STRIDE + pkh];
        char* brow = &BLds[buf][prow * B_STRIDE + pkh];
#pragma unroll
        for (int j = 0; j < 8; ++j) {
            v4i v = aReg[j];
            *(int*)(arow + 4 * j) = pack4(v[0], v[1], v[2], v[3]);
        }
#pragma unroll
        for (int j = 0; j < 8; ++j)
            *(int*)(brow + 4 * j) = pack4(bReg[4*j], bReg[4*j+1], bReg[4*j+2], bReg[4*j+3]);
    };

    load_global(0);
    pack_to_lds(0);

    int cur = 0;
    for (int k0 = 0; k0 < K_DIM; k0 += BK) {
        const bool more = (k0 + BK) < K_DIM;
        if (more) load_global(k0 + BK);

        __syncthreads();

        const char* abase0 = &ALds[cur][(mwave * 32 +  0 + l16) * A_STRIDE + half * 16];
        const char* abase1 = &ALds[cur][(mwave * 32 + 16 + l16) * A_STRIDE + half * 16];
        v8i a[2];
#pragma unroll
        for (int mi = 0; mi < 2; ++mi) {
            const char* base = mi ? abase1 : abase0;
            int2 p0 = *(const int2*)(base +  0);
            int2 p1 = *(const int2*)(base +  8);
            int2 p2 = *(const int2*)(base + 32);
            int2 p3 = *(const int2*)(base + 40);
            v8i av = {p0.x, p0.y, p1.x, p1.y, p2.x, p2.y, p3.x, p3.y};
            a[mi] = av;
        }

        v8i b[4];
#pragma unroll
        for (int ni = 0; ni < 4; ++ni) {
            const char* base = &BLds[cur][(nwave * 64 + ni * 16 + l16) * B_STRIDE + half * 16];
            int4 q0 = *(const int4*)(base +  0);
            int4 q1 = *(const int4*)(base + 32);
            v8i bv = {q0.x, q0.y, q0.z, q0.w, q1.x, q1.y, q1.z, q1.w};
            b[ni] = bv;
        }

#pragma unroll
        for (int mi = 0; mi < 2; ++mi)
#pragma unroll
            for (int ni = 0; ni < 4; ++ni)
                acc[mi][ni] = __builtin_amdgcn_wmma_i32_16x16x64_iu8(
                     true, a[mi],  true, b[ni],
                    acc[mi][ni],  false,  false);

        if (more) pack_to_lds(cur ^ 1);
        cur ^= 1;
    }

#pragma unroll
    for (int mi = 0; mi < 2; ++mi) {
#pragma unroll
        for (int ni = 0; ni < 4; ++ni) {
            const int   c  = nblk + nwave * 64 + ni * 16 + l16;
            const float bv = (float)bias[c];
            const int   r0 = mblk + mwave * 32 + mi * 16 + half * 8;
#pragma unroll
            for (int v = 0; v < 8; ++v) {
                float f = rintf((float)acc[mi][ni][v] * alpha + bv);
                f = fminf(fmaxf(f, -128.0f), 127.0f);
                __builtin_nontemporal_store(f, &out[(r0 + v) * O_DIM + c]);
            }
        }
    }
}

extern "C" void kernel_launch(void* const* d_in, const int* in_sizes, int n_in,
                              void* d_out, int out_size, void* d_ws, size_t ws_size,
                              hipStream_t stream) {
    const int*      x     = (const int*)d_in[0];
    const int*      w     = (const int*)d_in[1];
    const _Float16* bias  = (const _Float16*)d_in[2];
    const float*    alpha = (const float*)d_in[3];

    float* out = (float*)d_out;

    dim3 grid(N_ROWS / BM, O_DIM / BN);
    w8a8_gemm_wmma<<<grid, dim3(256), 0, stream>>>(x, w, bias, alpha, out);
}
